// LegacyDense_56006373540343
// MI455X (gfx1250) — matrix operand order repaired and checked by reading, not compiled, not run on hardware
//
#include <hip/hip_runtime.h>

typedef __attribute__((ext_vector_type(16))) _Float16 v16h;
typedef __attribute__((ext_vector_type(8)))  float    v8f;

#define B_DIM   64
#define T_DIM   512
#define IN_DIM  1024
#define OUT_DIM 1024

#define ALPHA 0.8187307530779818f   // exp(-dt/tau_syn)
#define BETA  0.9048374180359595f   // exp(-dt/tau_mem)

// ---------------------------------------------------------------------------
// Prep: w (IN x OUT, f32, row-major) -> wT (OUT x IN, f16, row-major).
// Makes each WMMA B-fragment a per-lane contiguous 32B load.
// ---------------------------------------------------------------------------
__global__ void transpose_w_f16(const float* __restrict__ w,
                                _Float16* __restrict__ wT) {
    __shared__ float tile[32][33];
    const int k = blockIdx.y * 32 + threadIdx.y;   // row of w
    const int o = blockIdx.x * 32 + threadIdx.x;   // col of w
    tile[threadIdx.y][threadIdx.x] = w[(size_t)k * OUT_DIM + o];
    __syncthreads();
    const int oo = blockIdx.x * 32 + threadIdx.y;  // row of wT
    const int kk = blockIdx.y * 32 + threadIdx.x;  // col of wT
    wT[(size_t)oo * IN_DIM + kk] = (_Float16)tile[threadIdx.x][threadIdx.y];
}

// ---------------------------------------------------------------------------
// Fused GEMM (f16 WMMA, f32 accum) + LIF scan.
// One wave per (batch b, 32-wide out-tile): two 16-wide N tiles share the A
// fragment (2 WMMAs per A load/convert). M = time, N = out-channel, so each
// accumulator holds a 16(T) x 16(OUT) tile of h:
//   acc[j]: lanes 0-15 -> t = t0+j, lanes 16-31 -> t = t0+j+8, N = lane%16.
// Scan state (syn, mem) per channel is replicated in both lane halves; the
// per-step spike store is branchless: lane L writes channel o0+L (cndmask
// picks spike0/spike1 by lane half) -> one coalesced 128B store per step,
// with compile-time immediate offsets off a per-tile base pointer.
// ---------------------------------------------------------------------------
__global__ __launch_bounds__(256)
void snn_fused_wmma(const float* __restrict__ x,
                    const _Float16* __restrict__ wT,
                    float* __restrict__ out) {
    const int lane  = threadIdx.x & 31;
    const int wave  = threadIdx.x >> 5;              // 0..7
    const int m     = lane & 15;                     // row-in-tile index
    const int half  = lane >> 4;                     // lane half
    const int b     = blockIdx.y;                    // batch
    const int o0    = (blockIdx.x * 8 + wave) * 32;  // 32-wide out-tile base

    // A fragment source rows (x, f32 -> f16 inline):
    // lane reads row t0+m, K = half*16 + [0..15].
    const size_t x_row0 = ((size_t)b * T_DIM) * IN_DIM;
    // B fragments (wT, f16): lane = channel, elems = K = half*16 + [0..15].
    const _Float16* wrow0 = wT + (size_t)(o0 + m) * IN_DIM + half * 16;
    const _Float16* wrow1 = wrow0 + (size_t)16 * IN_DIM;

    // Scan state for channels (o0+m) and (o0+16+m), replicated in both halves.
    float syn0 = 0.0f, mem0 = 0.0f;
    float syn1 = 0.0f, mem1 = 0.0f;

    for (int t0 = 0; t0 < T_DIM; t0 += 16) {
        v8f acc0 = {};
        v8f acc1 = {};

        const float* xp0 = x + x_row0 + (size_t)(t0 + m) * IN_DIM + half * 16;

#pragma unroll 2
        for (int k0 = 0; k0 < IN_DIM; k0 += 32) {
            // ---- A: 4x float4 loads + pack to f16 (co-executes with WMMA) ----
            const float4 f0 = *(const float4*)(xp0 + k0);
            const float4 f1 = *(const float4*)(xp0 + k0 + 4);
            const float4 f2 = *(const float4*)(xp0 + k0 + 8);
            const float4 f3 = *(const float4*)(xp0 + k0 + 12);
            v16h a;
            a[0]  = (_Float16)f0.x; a[1]  = (_Float16)f0.y;
            a[2]  = (_Float16)f0.z; a[3]  = (_Float16)f0.w;
            a[4]  = (_Float16)f1.x; a[5]  = (_Float16)f1.y;
            a[6]  = (_Float16)f1.z; a[7]  = (_Float16)f1.w;
            a[8]  = (_Float16)f2.x; a[9]  = (_Float16)f2.y;
            a[10] = (_Float16)f2.z; a[11] = (_Float16)f2.w;
            a[12] = (_Float16)f3.x; a[13] = (_Float16)f3.y;
            a[14] = (_Float16)f3.z; a[15] = (_Float16)f3.w;

            // ---- B: one contiguous 32B per-lane load per N tile ----
            const v16h bf0 = *(const v16h*)(wrow0 + k0);
            const v16h bf1 = *(const v16h*)(wrow1 + k0);

            // Two 16x16x32 WMMAs sharing the A fragment.
            acc0 = __builtin_amdgcn_wmma_f32_16x16x32_f16(
                false, a, false, bf0, (short)0, acc0, false, false);
            acc1 = __builtin_amdgcn_wmma_f32_16x16x32_f16(
                false, a, false, bf1, (short)0, acc1, false, false);
        }

        // ---- Sequential LIF scan over the 16 time steps of this tile ----
        // t0+0..7 live in lane-half 0 of acc[0..7], t0+8..15 in half 1.
        float* op = out + ((size_t)(b * T_DIM + t0)) * OUT_DIM + o0 + lane;

#pragma unroll
        for (int ph = 0; ph < 2; ++ph) {
#pragma unroll
            for (int j = 0; j < 8; ++j) {
                // Broadcast h(t, channel) to both lane halves.
                const float h0 = __shfl(acc0[j], m + ph * 16, 32);
                const float h1 = __shfl(acc1[j], m + ph * 16, 32);

                const float s0 = (mem0 - 1.0f > 0.0f) ? 1.0f : 0.0f;
                const float s1 = (mem1 - 1.0f > 0.0f) ? 1.0f : 0.0f;
                const float nsyn0 = ALPHA * syn0 + h0;
                const float nsyn1 = ALPHA * syn1 + h1;
                const float nmem0 = BETA * mem0 + syn0 - s0;  // rst == spike
                const float nmem1 = BETA * mem1 + syn1 - s1;

                // Lane L owns channel o0+L: half 0 -> s0, half 1 -> s1.
                // One coalesced 128B store, immediate offset per step.
                op[(size_t)(ph * 8 + j) * OUT_DIM] = half ? s1 : s0;

                syn0 = nsyn0; mem0 = nmem0;
                syn1 = nsyn1; mem1 = nmem1;
            }
        }
    }
}

// ---------------------------------------------------------------------------
extern "C" void kernel_launch(void* const* d_in, const int* in_sizes, int n_in,
                              void* d_out, int out_size, void* d_ws, size_t ws_size,
                              hipStream_t stream) {
    const float* x = (const float*)d_in[0];   // (64, 512, 1024) f32
    const float* w = (const float*)d_in[1];   // (1024, 1024) f32
    float* out     = (float*)d_out;           // (64, 512, 1024) f32
    _Float16* wT   = (_Float16*)d_ws;         // 2 MB scratch: wT (OUT x IN) f16

    // 1) Transpose + downconvert weights (one-time, 4 MB read / 2 MB write).
    transpose_w_f16<<<dim3(OUT_DIM / 32, IN_DIM / 32), dim3(32, 32), 0, stream>>>(w, wT);

    // 2) Fused WMMA GEMM + LIF scan: 256 blocks x 8 waves = 2048 waves,
    //    each wave owns (batch, 32-channel tile) for all 512 time steps.
    snn_fused_wmma<<<dim3(OUT_DIM / (32 * 8), B_DIM), 256, 0, stream>>>(x, wT, out);
}
